// TernaryDense_21182778703910
// MI455X (gfx1250) — compile-verified, matrix operand order checked by reading, not run on hardware
//
#include <hip/hip_runtime.h>

// ---------------------------------------------------------------------------
// TernaryDense forward for MI455X (gfx1250, wave32).
//   C[N,U] = A[N,D] (f32) @ ternarize(W[D,U]),  ternarize = rint(clip(w,-1,1))
//
// Strategy: split-f16 activations (x = hi + lo, both f16) x exact-f16 ternary
// weights, f32 accumulation via v_wmma_f32_16x16x32_f16.
//
//  - PRE=true : pre-pass converts A -> (Ahi,Alo) f16 and W -> ternarize(W)^T
//               f16 [U][D]; lean GEMM stages tiles with CDNA5 async
//               global->LDS copies (ASYNCcnt) when available, else via VGPRs.
//  - PRE=false: fully fused conversion + transpose inside the GEMM.
// ---------------------------------------------------------------------------

#if defined(__HIP_DEVICE_COMPILE__) && __has_builtin(__builtin_amdgcn_global_load_async_to_lds_b128)
#define HAS_ASYNC_LDS 1
#else
#define HAS_ASYNC_LDS 0
#endif

typedef _Float16 v8h  __attribute__((ext_vector_type(8)));
typedef _Float16 v16h __attribute__((ext_vector_type(16)));
typedef float    v8f  __attribute__((ext_vector_type(8)));

#define N_DIM 8192
#define D_DIM 4096
#define U_DIM 4096

#define BM 128
#define BN 128
#define BK 32
#define LDS_STRIDE 40          // 32 f16 + 8 pad -> 80 B/row, 16B aligned
#define K_TILES (D_DIM / BK)   // 128

__device__ __forceinline__ _Float16 ternarize(float x) {
    float c = fminf(1.0f, fmaxf(-1.0f, x));
    return (_Float16)rintf(c);          // round-half-to-even == jnp.round
}

// Load one 16x32 f16 fragment (per-lane: K[kb..kb+8) then K[kb+16..kb+24))
__device__ __forceinline__ v16h load_frag(const _Float16* base) {
    v8h p0 = *(const v8h*)(base);
    v8h p1 = *(const v8h*)(base + 16);
    return __builtin_shufflevector(p0, p1,
                                   0, 1, 2, 3, 4, 5, 6, 7,
                                   8, 9, 10, 11, 12, 13, 14, 15);
}

__device__ __forceinline__ v8f wmma_f16(v16h a, v16h b, v8f c) {
    return __builtin_amdgcn_wmma_f32_16x16x32_f16(
        /*neg_a=*/false, a, /*neg_b=*/false, b,
        /*c_mod=*/(short)0, c, /*reuse_a=*/false, /*reuse_b=*/false);
}

#if HAS_ASYNC_LDS
typedef int v4i __attribute__((vector_size(16)));
// 16-byte async DMA: global -> LDS, tracked by ASYNCcnt.
__device__ __forceinline__ void async_copy16(const void* g, void* l) {
    __builtin_amdgcn_global_load_async_to_lds_b128(
        (__attribute__((address_space(1))) v4i*)g,
        (__attribute__((address_space(3))) v4i*)l,
        /*imm offset=*/0, /*cpol=*/0);
}
#endif

// ---------------------------------------------------------------------------
// Pre-pass kernels (memory bound)
// ---------------------------------------------------------------------------
__global__ __launch_bounds__(256)
void prep_a_kernel(const float* __restrict__ A,
                   _Float16* __restrict__ Ahi, _Float16* __restrict__ Alo) {
    const size_t base = ((size_t)blockIdx.x * 256 + threadIdx.x) * 8;
    const float4* p = (const float4*)(A + base);
    const float4 q0 = p[0], q1 = p[1];
    const float xs[8] = {q0.x, q0.y, q0.z, q0.w, q1.x, q1.y, q1.z, q1.w};
    v8h h{}, l{};
#pragma unroll
    for (int i = 0; i < 8; ++i) {
        const _Float16 hh = (_Float16)xs[i];
        h[i] = hh;
        l[i] = (_Float16)(xs[i] - (float)hh);
    }
    *(v8h*)(Ahi + base) = h;
    *(v8h*)(Alo + base) = l;
}

// Ternarize + transpose: W[D,U] f32 -> Wt[U,D] f16, 32x32 tiles through LDS.
__global__ __launch_bounds__(256)
void prep_wt_kernel(const float* __restrict__ W, _Float16* __restrict__ Wt) {
    __shared__ _Float16 tile[32][33];   // +1 pad breaks bank conflicts
    const int tx = threadIdx.x & 31;
    const int ty = threadIdx.x >> 5;    // 0..7
    const int k0 = blockIdx.y * 32;
    const int u0 = blockIdx.x * 32;
#pragma unroll
    for (int s = 0; s < 4; ++s) {       // coalesced read along U
        const int k = ty + s * 8;
        tile[k][tx] = ternarize(W[(size_t)(k0 + k) * U_DIM + u0 + tx]);
    }
    __syncthreads();
#pragma unroll
    for (int s = 0; s < 4; ++s) {       // coalesced write along D
        const int u = ty + s * 8;
        Wt[(size_t)(u0 + u) * D_DIM + k0 + tx] = tile[tx][u];
    }
}

// ---------------------------------------------------------------------------
// Main GEMM (templated: PRE=true reads pre-converted f16, PRE=false fuses)
// ---------------------------------------------------------------------------
template <bool PRE>
__global__ __launch_bounds__(256, 1)
void ternary_gemm(const float* __restrict__ A32,
                  const float* __restrict__ W32,
                  const _Float16* __restrict__ Ahi,
                  const _Float16* __restrict__ Alo,
                  const _Float16* __restrict__ Wt,   // PRE: [U][D] f16
                  float* __restrict__ C) {
    __shared__ _Float16 sAhi[2][BM * LDS_STRIDE];
    __shared__ _Float16 sAlo[2][BM * LDS_STRIDE];
    __shared__ _Float16 sBt [2][BN * LDS_STRIDE];   // W tile, N-major rows

    const int tid  = threadIdx.x;
    const int lane = tid & 31;
    const int wave = tid >> 5;

    const int blockM = blockIdx.y * BM;
    const int blockN = blockIdx.x * BN;

    // A/B-tile loader: one thread = 16 contiguous elements of one tile row
    const int ar = tid >> 1;            // 0..127 : row within tile
    const int ah = (tid & 1) * 16;      // 0 / 16 : K offset within BK

    // Fused-path W loader: one thread = 2 k-rows x 8 cols
    const int wk2 = (tid >> 4) * 2;     // 0..30 (even) : k-pair within tile
    const int wcg = (tid & 15) * 8;     // 0..120       : col group within BN

    // Wave tiling: 8 waves in 2 (M) x 4 (N); each wave 64x32 of C
    const int wm = (wave >> 2) * 64;
    const int wn = (wave & 3) * 32;
    const int fm = lane & 15;
    const int kb = (lane >> 4) * 8;

    v8f acc[4][2];
#pragma unroll
    for (int mt = 0; mt < 4; ++mt)
#pragma unroll
        for (int nt = 0; nt < 2; ++nt)
            acc[mt][nt] = (v8f){0.f, 0.f, 0.f, 0.f, 0.f, 0.f, 0.f, 0.f};

    // Staging registers (register-staged paths only)
    float4 aReg[4]; float4 wReg[4];         // fused path
    v8h pAhi[2], pAlo[2], pB[2];            // pre-pass, no-async fallback

    auto stage_global = [&](int kt) {
        if constexpr (PRE) {
            const size_t aoff = (size_t)(blockM + ar) * D_DIM + kt * BK + ah;
            pAhi[0] = ((const v8h*)(Ahi + aoff))[0];
            pAhi[1] = ((const v8h*)(Ahi + aoff))[1];
            pAlo[0] = ((const v8h*)(Alo + aoff))[0];
            pAlo[1] = ((const v8h*)(Alo + aoff))[1];
            const _Float16* wp = Wt + (size_t)(blockN + ar) * D_DIM + kt * BK + ah;
            pB[0] = ((const v8h*)wp)[0];
            pB[1] = ((const v8h*)wp)[1];
        } else {
            const float* ap = A32 + (size_t)(blockM + ar) * D_DIM + kt * BK + ah;
            const float4* av = (const float4*)ap;
            aReg[0] = av[0]; aReg[1] = av[1]; aReg[2] = av[2]; aReg[3] = av[3];
            const float* wp0 = W32 + (size_t)(kt * BK + wk2) * U_DIM + blockN + wcg;
            const float4* wv0 = (const float4*)wp0;
            const float4* wv1 = (const float4*)(wp0 + U_DIM);
            wReg[0] = wv0[0]; wReg[1] = wv0[1];
            wReg[2] = wv1[0]; wReg[3] = wv1[1];
        }
    };

    auto commit_lds = [&](int buf) {
        if constexpr (PRE) {
            _Float16* dH = &sAhi[buf][ar * LDS_STRIDE + ah];
            *(v8h*)dH = pAhi[0]; *(v8h*)(dH + 8) = pAhi[1];
            _Float16* dL = &sAlo[buf][ar * LDS_STRIDE + ah];
            *(v8h*)dL = pAlo[0]; *(v8h*)(dL + 8) = pAlo[1];
            _Float16* dB = &sBt[buf][ar * LDS_STRIDE + ah];
            *(v8h*)dB = pB[0]; *(v8h*)(dB + 8) = pB[1];
        } else {
            v8h h0{}, h1{}, l0{}, l1{};
#pragma unroll
            for (int i = 0; i < 4; ++i) {
                const float xs[4] = {aReg[i].x, aReg[i].y, aReg[i].z, aReg[i].w};
#pragma unroll
                for (int j = 0; j < 4; ++j) {
                    const int idx = i * 4 + j;
                    const float x = xs[j];
                    const _Float16 h = (_Float16)x;
                    const _Float16 l = (_Float16)(x - (float)h);
                    if (idx < 8) { h0[idx] = h; l0[idx] = l; }
                    else         { h1[idx - 8] = h; l1[idx - 8] = l; }
                }
            }
            _Float16* dH = &sAhi[buf][ar * LDS_STRIDE + ah];
            *(v8h*)dH = h0; *(v8h*)(dH + 8) = h1;
            _Float16* dL = &sAlo[buf][ar * LDS_STRIDE + ah];
            *(v8h*)dL = l0; *(v8h*)(dL + 8) = l1;

            const float r0[8] = {wReg[0].x, wReg[0].y, wReg[0].z, wReg[0].w,
                                 wReg[1].x, wReg[1].y, wReg[1].z, wReg[1].w};
            const float r1[8] = {wReg[2].x, wReg[2].y, wReg[2].z, wReg[2].w,
                                 wReg[3].x, wReg[3].y, wReg[3].z, wReg[3].w};
#pragma unroll
            for (int j = 0; j < 8; ++j) {
                union { _Float16 h[2]; unsigned u; } pk;
                pk.h[0] = ternarize(r0[j]);
                pk.h[1] = ternarize(r1[j]);
                *(unsigned*)&sBt[buf][(wcg + j) * LDS_STRIDE + wk2] = pk.u;
            }
        }
    };

    auto compute = [&](int buf) {
        // Hold only the 2 B-fragments; stream one (hi,lo) A-pair per m-tile.
        v16h bfr[2];
#pragma unroll
        for (int nt = 0; nt < 2; ++nt)
            bfr[nt] = load_frag(&sBt[buf][(wn + nt * 16 + fm) * LDS_STRIDE + kb]);
#pragma unroll
        for (int mt = 0; mt < 4; ++mt) {
            const int ab = (wm + mt * 16 + fm) * LDS_STRIDE + kb;
            const v16h fhi = load_frag(&sAhi[buf][ab]);
            const v16h flo = load_frag(&sAlo[buf][ab]);
#pragma unroll
            for (int nt = 0; nt < 2; ++nt) {
                acc[mt][nt] = wmma_f16(fhi, bfr[nt], acc[mt][nt]);
                acc[mt][nt] = wmma_f16(flo, bfr[nt], acc[mt][nt]);
            }
        }
    };

    auto run_staged = [&]() {
        stage_global(0);
        commit_lds(0);
        __syncthreads();
        for (int kt = 0; kt < K_TILES; ++kt) {
            const int cur = kt & 1;
            if (kt + 1 < K_TILES) stage_global(kt + 1);
            compute(cur);
            if (kt + 1 < K_TILES) commit_lds(cur ^ 1);
            __syncthreads();
        }
    };

    if constexpr (PRE) {
#if HAS_ASYNC_LDS
        // Async DMA staging: issue copies for tile kt+1 into the other buffer,
        // run WMMAs on tile kt, then a single asynccnt wait + barrier.
        auto stage_async = [&](int kt, int buf) {
            const size_t aoff = (size_t)(blockM + ar) * D_DIM + kt * BK + ah;
            const size_t boff = (size_t)(blockN + ar) * D_DIM + kt * BK + ah;
            _Float16* dH = &sAhi[buf][ar * LDS_STRIDE + ah];
            _Float16* dL = &sAlo[buf][ar * LDS_STRIDE + ah];
            _Float16* dB = &sBt [buf][ar * LDS_STRIDE + ah];
            async_copy16(Ahi + aoff,     dH);
            async_copy16(Ahi + aoff + 8, dH + 8);
            async_copy16(Alo + aoff,     dL);
            async_copy16(Alo + aoff + 8, dL + 8);
            async_copy16(Wt  + boff,     dB);
            async_copy16(Wt  + boff + 8, dB + 8);
        };

        stage_async(0, 0);
        asm volatile("s_wait_asynccnt 0" ::: "memory");
        __syncthreads();
        // unroll 1: a single loop body forces the accumulators to live in the
        // same physical registers at the back-edge (no acc shuffling / NOPs).
#pragma unroll 1
        for (int kt = 0; kt < K_TILES; ++kt) {
            const int cur = kt & 1;
            if (kt + 1 < K_TILES) stage_async(kt + 1, cur ^ 1);
            compute(cur);
            if (kt + 1 < K_TILES) asm volatile("s_wait_asynccnt 0" ::: "memory");
            __syncthreads();
        }
#else
        run_staged();
#endif
    } else {
        run_staged();
    }

    // ---- epilogue: C/D layout (VGPR r: M = r + 8*(lane>>4), N = lane&15) ----
#pragma unroll
    for (int mt = 0; mt < 4; ++mt) {
        const int gm = blockM + wm + mt * 16 + (lane >> 4) * 8;
#pragma unroll
        for (int nt = 0; nt < 2; ++nt) {
            const int gn = blockN + wn + nt * 16 + (lane & 15);
            float* cp = C + (size_t)gm * U_DIM + gn;
#pragma unroll
            for (int r = 0; r < 8; ++r)
                cp[(size_t)r * U_DIM] = acc[mt][nt][r];
        }
    }
}

// ---------------------------------------------------------------------------
extern "C" void kernel_launch(void* const* d_in, const int* in_sizes, int n_in,
                              void* d_out, int out_size, void* d_ws, size_t ws_size,
                              hipStream_t stream) {
    const float* A = (const float*)d_in[0];   // inputs [8192, 4096] f32
    const float* W = (const float*)d_in[1];   // shadow weights [4096, 4096] f32
    float* C = (float*)d_out;                 // output [8192, 4096] f32

    const size_t aBytes = (size_t)N_DIM * D_DIM * sizeof(_Float16); // 64 MiB
    const size_t wBytes = (size_t)D_DIM * U_DIM * sizeof(_Float16); // 32 MiB
    const size_t need   = 2 * aBytes + wBytes;                      // 160 MiB

    dim3 grid(U_DIM / BN, N_DIM / BM);        // 32 x 64 blocks of 256 threads

    if (d_ws != nullptr && ws_size >= need) {
        _Float16* Ahi = (_Float16*)d_ws;
        _Float16* Alo = (_Float16*)((char*)d_ws + aBytes);
        _Float16* Wt  = (_Float16*)((char*)d_ws + 2 * aBytes);

        const int aBlocks = (int)(((size_t)N_DIM * D_DIM) / (8 * 256)); // 16384
        prep_a_kernel<<<aBlocks, 256, 0, stream>>>(A, Ahi, Alo);
        dim3 wtGrid(U_DIM / 32, D_DIM / 32);                            // 128x128
        prep_wt_kernel<<<wtGrid, 256, 0, stream>>>(W, Wt);
        ternary_gemm<true><<<grid, 256, 0, stream>>>(A, W, Ahi, Alo, Wt, C);
    } else {
        ternary_gemm<false><<<grid, 256, 0, stream>>>(A, W, nullptr, nullptr, nullptr, C);
    }
}
